// QSVDDModel_34686155882825
// MI455X (gfx1250) — matrix operand order repaired and checked by reading, not compiled, not run on hardware
//
#include <hip/hip_runtime.h>
#include <hip/hip_bf16.h>

typedef __attribute__((ext_vector_type(16))) __bf16 v16bf;
typedef __attribute__((ext_vector_type(8)))  float  v8f;

union Frag {
    v16bf v;
    uint4 q[2];
};

struct cplx { float re, im; };

__device__ __forceinline__ cplx cmul(cplx a, cplx b) {
    return {a.re * b.re - a.im * b.im, a.re * b.im + a.im * b.re};
}

__device__ __forceinline__ unsigned short f2bf(float f) {
    unsigned int u = __float_as_uint(f);
    u += 0x7fffu + ((u >> 16) & 1u);           // round-to-nearest-even
    return (unsigned short)(u >> 16);
}

// ---------- 4x4 SU(4) block construction (matches reference _su4) ----------
__device__ void mk_u3(float t, float p, float l, cplx* u) {
    float ct = cosf(0.5f * t), st = sinf(0.5f * t);
    u[0] = {ct, 0.f};
    u[1] = {-cosf(l) * st, -sinf(l) * st};
    u[2] = {cosf(p) * st, sinf(p) * st};
    u[3] = {cosf(p + l) * ct, sinf(p + l) * ct};
}
__device__ void mk_ry(float t, cplx* u) {
    float c = cosf(0.5f * t), s = sinf(0.5f * t);
    u[0] = {c, 0.f}; u[1] = {-s, 0.f}; u[2] = {s, 0.f}; u[3] = {c, 0.f};
}
__device__ void mk_rz(float t, cplx* u) {
    float c = cosf(0.5f * t), s = sinf(0.5f * t);
    u[0] = {c, -s}; u[1] = {0.f, 0.f}; u[2] = {0.f, 0.f}; u[3] = {c, s};
}
__device__ void mk_id(cplx* u) {
    u[0] = {1.f, 0.f}; u[1] = {0.f, 0.f}; u[2] = {0.f, 0.f}; u[3] = {1.f, 0.f};
}
__device__ void kron4(const cplx* A, const cplx* B, cplx* M) {
    for (int r = 0; r < 4; ++r)
        for (int c = 0; c < 4; ++c)
            M[r * 4 + c] = cmul(A[(r >> 1) * 2 + (c >> 1)], B[(r & 1) * 2 + (c & 1)]);
}
__device__ void mm4(const cplx* A, const cplx* B, cplx* C) {
    for (int r = 0; r < 4; ++r)
        for (int c = 0; c < 4; ++c) {
            cplx s = {0.f, 0.f};
            for (int t = 0; t < 4; ++t) {
                cplx p = cmul(A[r * 4 + t], B[t * 4 + c]);
                s.re += p.re; s.im += p.im;
            }
            C[r * 4 + c] = s;
        }
}
__device__ void swap_rows(cplx* M, int r0, int r1) {
    for (int c = 0; c < 4; ++c) {
        cplx t = M[r0 * 4 + c]; M[r0 * 4 + c] = M[r1 * 4 + c]; M[r1 * 4 + c] = t;
    }
}
__device__ void su4(const float* p, cplx* M) {
    cplx u0[4], u1[4], K[16], T[16];
    mk_u3(p[0], p[1], p[2], u0); mk_u3(p[3], p[4], p[5], u1);
    kron4(u0, u1, M);
    swap_rows(M, 2, 3);                                   // CNOT01 @ M
    mk_ry(p[6], u0); mk_rz(p[7], u1);
    kron4(u0, u1, K); mm4(K, M, T);
    for (int i = 0; i < 16; ++i) M[i] = T[i];
    swap_rows(M, 1, 3);                                   // CNOT10 @ M
    mk_ry(p[8], u0); mk_id(u1);
    kron4(u0, u1, K); mm4(K, M, T);
    for (int i = 0; i < 16; ++i) M[i] = T[i];
    swap_rows(M, 2, 3);                                   // CNOT01 @ M
    mk_u3(p[9], p[10], p[11], u0); mk_u3(p[12], p[13], p[14], u1);
    kron4(u0, u1, K); mm4(K, M, T);
    for (int i = 0; i < 16; ++i) M[i] = T[i];
}

// ---------- K1: evolve each basis column through the 25-gate circuit ----------
// Ucol[j*256 + i] = U[i][j]  (column-major storage of the full circuit unitary)
__global__ __launch_bounds__(64) void build_unitary(const float* __restrict__ qw,
                                                    float2* __restrict__ Ucol) {
    __shared__ cplx col[256];
    __shared__ cplx Mg[5][16];
    const int j = blockIdx.x, t = threadIdx.x;
    if (t < 5) su4(qw + t * 15, &Mg[t][0]);
    for (int i = t; i < 256; i += 64) { col[i].re = (i == j) ? 1.f : 0.f; col[i].im = 0.f; }
    __syncthreads();

    const int GL[25] = {0,0,0,0,0,0,0,0, 1,1,1,1,1,1,1,1, 2,2,2,2, 3,3,3,3, 4};
    const int GA[25] = {0,2,4,6,1,3,5,7, 0,2,4,6,1,3,5,7, 2,6,0,4, 2,6,0,4, 2};
    const int GB[25] = {1,3,5,7,2,4,6,0, 1,3,5,7,2,4,6,0, 4,0,2,6, 4,0,2,6, 6};

#pragma unroll
    for (int g = 0; g < 25; ++g) {
        const int b0 = 7 - GA[g], b1 = 7 - GB[g];          // wire w -> bit 7-w
        const int lo = (b0 < b1) ? b0 : b1;
        const int hi = (b0 < b1) ? b1 : b0;
        const cplx* M = &Mg[GL[g]][0];
        // each of 64 threads owns one disjoint 4-amplitude group
        int base = ((t >> lo) << (lo + 1)) | (t & ((1 << lo) - 1));
        base = ((base >> hi) << (hi + 1)) | (base & ((1 << hi) - 1));
        const int i0 = base;
        const int i1 = base | (1 << b1);
        const int i2 = base | (1 << b0);
        const int i3 = base | (1 << b0) | (1 << b1);
        cplx o[4] = {col[i0], col[i1], col[i2], col[i3]};
        cplx n[4];
#pragma unroll
        for (int r = 0; r < 4; ++r) {
            cplx s = {0.f, 0.f};
#pragma unroll
            for (int c = 0; c < 4; ++c) {
                cplx p = cmul(M[r * 4 + c], o[c]);
                s.re += p.re; s.im += p.im;
            }
            n[r] = s;
        }
        col[i0] = n[0]; col[i1] = n[1]; col[i2] = n[2]; col[i3] = n[3];
        __syncthreads();
    }
    for (int i = t; i < 256; i += 64) {
        float2 v; v.x = col[i].re; v.y = col[i].im;
        Ucol[j * 256 + i] = v;
    }
}

// ---------- K2: S_k = Re(U^dag P_k U) in bf16, k = {X,Y,Z}@wire2, {X,Y,Z}@wire6 ----------
__global__ __launch_bounds__(256) void build_smats(const float2* __restrict__ Ucol,
                                                   unsigned short* __restrict__ Sbf) {
    const int i = blockIdx.x;
    const int wq = blockIdx.y;               // 0 -> wire2 (bit 5), 1 -> wire6 (bit 1)
    const int mask = wq ? 2 : 32;
    __shared__ float2 colI[256];
    const int j = threadIdx.x;
    colI[j] = Ucol[i * 256 + j];             // U[a][i], contiguous in a
    __syncthreads();
    const float2* myc = Ucol + j * 256;      // U[.][j]
    float ax = 0.f, ay = 0.f, az = 0.f;
    for (int a = 0; a < 256; ++a) {
        float2 u = colI[a];                  // U[a][i]
        float2 v = myc[a ^ mask];            // U[a^m][j]
        float2 w = myc[a];                   // U[a][j]
        float c1re = u.x * v.x + u.y * v.y;  // Re(conj(u)*v)
        float c1im = u.x * v.y - u.y * v.x;  // Im(conj(u)*v)
        float c2re = u.x * w.x + u.y * w.y;  // Re(conj(u)*w)
        float sgn = (a & mask) ? -1.f : 1.f;
        ax += c1re;                          // X
        ay += sgn * c1im;                    // Y: +Im if bit=0, -Im if bit=1
        az += sgn * c2re;                    // Z
    }
    const int kb = wq * 3;
    Sbf[(kb + 0) * 65536 + i * 256 + j] = f2bf(ax);
    Sbf[(kb + 1) * 65536 + i * 256 + j] = f2bf(ay);
    Sbf[(kb + 2) * 65536 + i * 256 + j] = f2bf(az);
}

// ---------- K3: out[b,k] = (x_b^T S_k x_b) / ||x_b||^2 via WMMA bf16 ----------
// block = 192 threads (6 waves, wave k owns S_k), 32 batch rows per block.
__global__ __launch_bounds__(192) void qforms(const float* __restrict__ X,
                                              const unsigned short* __restrict__ Sbf,
                                              float* __restrict__ out, int B) {
    __shared__ __align__(16) float          Xf[32 * 256];   // 32 KB f32 tile
    __shared__ __align__(16) unsigned short Xb[32 * 256];   // 16 KB bf16 tile
    __shared__ float nrm[32];
    __shared__ float part[6][2][32][8];                     // 12 KB partials

    const int tid = threadIdx.x;
    const int b0  = blockIdx.x * 32;

    for (int idx = tid; idx < 32 * 256; idx += 192) {
        int r = idx >> 8, c = idx & 255;
        float v = (b0 + r < B) ? X[(b0 + r) * 256 + c] : 0.f;
        Xf[idx] = v;
        Xb[idx] = f2bf(v);
    }
    __syncthreads();
    if (tid < 32) {                                         // deterministic row norms
        float s = 0.f;
        for (int c = 0; c < 256; ++c) { float v = Xf[tid * 256 + c]; s += v * v; }
        nrm[tid] = s;
    }

    const int wave = tid >> 5;
    const int lane = tid & 31;
    const int nl   = lane & 15;
    const int hi   = lane >> 4;
    const unsigned short* Sk = Sbf + wave * 65536;

    float p0[8], p1[8];
#pragma unroll
    for (int v = 0; v < 8; ++v) { p0[v] = 0.f; p1[v] = 0.f; }

    for (int nt = 0; nt < 16; ++nt) {
        v8f acc0 = (v8f)0.0f;
        v8f acc1 = (v8f)0.0f;
        const int n = nt * 16 + nl;
#pragma unroll
        for (int ks = 0; ks < 8; ++ks) {
            Frag bfr, a0, a1;
            // B frag (S is symmetric: row n == column n): lanes 0-15 K {0-7,16-23}, lanes 16-31 K {8-15,24-31}
            const unsigned short* pb = Sk + n * 256 + ks * 32 + hi * 8;
            bfr.q[0] = *(const uint4*)pb;
            bfr.q[1] = *(const uint4*)(pb + 16);
            // A frags: lanes 0-15 K {0-7,16-23}, lanes 16-31 K {8-15,24-31}
            const int ka = ks * 32 + hi * 8;
            const unsigned short* pa0 = &Xb[nl * 256 + ka];
            a0.q[0] = *(const uint4*)pa0;
            a0.q[1] = *(const uint4*)(pa0 + 16);
            const unsigned short* pa1 = &Xb[(16 + nl) * 256 + ka];
            a1.q[0] = *(const uint4*)pa1;
            a1.q[1] = *(const uint4*)(pa1 + 16);

            acc0 = __builtin_amdgcn_wmma_f32_16x16x32_bf16(false, a0.v, false, bfr.v,
                                                           (short)0, acc0, false, false);
            acc1 = __builtin_amdgcn_wmma_f32_16x16x32_bf16(false, a1.v, false, bfr.v,
                                                           (short)0, acc1, false, false);
        }
        // fused epilogue: partial += X[m,n] * (X S)[m,n]; C layout: VGPR v -> M = v (+8 for hi lanes)
#pragma unroll
        for (int v = 0; v < 8; ++v) {
            const int m = hi ? v + 8 : v;
            p0[v] += Xf[m * 256 + n] * acc0[v];
            p1[v] += Xf[(16 + m) * 256 + n] * acc1[v];
        }
    }
#pragma unroll
    for (int v = 0; v < 8; ++v) {
        part[wave][0][lane][v] = p0[v];
        part[wave][1][lane][v] = p1[v];
    }
    __syncthreads();
    if (lane < 16) {
        const int m  = lane;
        const int v  = m & 7;
        const int lo = (m >= 8) ? 16 : 0;
#pragma unroll
        for (int mt = 0; mt < 2; ++mt) {
            float s = 0.f;
            for (int l = 0; l < 16; ++l) s += part[wave][mt][lo + l][v];
            const int row = b0 + mt * 16 + m;
            if (row < B) out[row * 6 + wave] = s / nrm[mt * 16 + m];
        }
    }
}

extern "C" void kernel_launch(void* const* d_in, const int* in_sizes, int n_in,
                              void* d_out, int out_size, void* d_ws, size_t ws_size,
                              hipStream_t stream) {
    const float* X  = (const float*)d_in[0];   // [B, 256] f32
    const float* qw = (const float*)d_in[1];   // [75] f32
    const int B = in_sizes[0] / 256;

    float2* Ucol = (float2*)d_ws;                                        // 512 KB
    unsigned short* Sbf = (unsigned short*)((char*)d_ws + 256 * 256 * sizeof(float2)); // 768 KB

    build_unitary<<<256, 64, 0, stream>>>(qw, Ucol);
    build_smats<<<dim3(256, 2), 256, 0, stream>>>(Ucol, Sbf);
    qforms<<<(B + 31) / 32, 192, 0, stream>>>(X, Sbf, (float*)d_out, B);
}
